// MultimodalAttention_25074019074500
// MI455X (gfx1250) — matrix operand order repaired and checked by reading, not compiled, not run on hardware
//
#include <hip/hip_runtime.h>
#include <hip/hip_bf16.h>
#include <math.h>

// ---------------------------------------------------------------------------
// Problem constants (reference: EMB=1024, HEADS=16, B=2, I=1024, CT=512)
// ---------------------------------------------------------------------------
#define EMBD 1024
#define NHEADS 16
#define HSZ 64
#define NB 2
#define SEQI 1024
#define SEQC 512
#define JTOT 2048 // I + 2*CT

typedef __attribute__((ext_vector_type(16))) __bf16 v16bf;
typedef __attribute__((ext_vector_type(16))) unsigned short v16us;
typedef __attribute__((ext_vector_type(8))) unsigned short v8us;
typedef __attribute__((ext_vector_type(8))) float v8f;

union FragAB {
  v16us u;
  v8us h[2]; // h[0] = elements 0..7, h[1] = elements 8..15
  v16bf b;
};

// f32 -> bf16, round-to-nearest (hi16 of x + 0x8000). With a ds/global b16
// store of the value the compiler folds the >>16 into a *_d16_hi store.
__device__ __forceinline__ unsigned short f2bf(float f) {
  return (unsigned short)((__builtin_bit_cast(unsigned int, f) + 0x8000u) >>
                          16);
}

// Packed f32x2 -> bf16x2: hardware cvt if declared, else 2 adds + 1 v_perm.
__device__ __forceinline__ unsigned int f2bf2(float lo, float hi) {
#if __has_builtin(__builtin_amdgcn_cvt_pk_bf16_f32)
  auto r = __builtin_amdgcn_cvt_pk_bf16_f32(lo, hi);
  return __builtin_bit_cast(unsigned int, r);
#else
  unsigned int a = __builtin_bit_cast(unsigned int, lo) + 0x8000u;
  unsigned int b = __builtin_bit_cast(unsigned int, hi) + 0x8000u;
  return __builtin_amdgcn_perm(b, a, 0x07060302u); // {b[31:16], a[31:16]}
#endif
}

// exp2 on the trans unit (v_exp_f32 is base-2 on AMD)
__device__ __forceinline__ float fast_exp2(float x) {
#if __has_builtin(__builtin_amdgcn_exp2f)
  return __builtin_amdgcn_exp2f(x);
#else
  return exp2f(x);
#endif
}

// Async global->LDS 128-bit copy (CDNA5, ASYNCcnt-tracked, GV addressing).
// LDS destination address = low 32 bits of the generic shared pointer
// (flat->LDS truncation per ISA 10.2).
__device__ __forceinline__ void async_b128(void* lds, const void* g) {
  unsigned off = (unsigned)(uintptr_t)lds;
  unsigned long long ga = (unsigned long long)(uintptr_t)g;
  asm volatile("global_load_async_to_lds_b128 %0, %1, off" ::"v"(off), "v"(ga)
               : "memory");
}
__device__ __forceinline__ void wait_async0() {
  asm volatile("s_wait_asynccnt 0" ::: "memory");
}

// ---------------------------------------------------------------------------
// Concatenate [mask | mm_mask0 | mm_mask1] -> kmask[B][JTOT]
// ---------------------------------------------------------------------------
__global__ void build_mask(const float* __restrict__ m,
                           const float* __restrict__ mm0,
                           const float* __restrict__ mm1,
                           float* __restrict__ outm) {
  int idx = blockIdx.x * blockDim.x + threadIdx.x;
  if (idx >= NB * JTOT) return;
  int b = idx / JTOT, j = idx % JTOT;
  float v;
  if (j < SEQI) v = m[b * SEQI + j];
  else if (j < SEQI + SEQC) v = mm0[b * SEQC + (j - SEQI)];
  else v = mm1[b * SEQC + (j - SEQI - SEQC)];
  outm[idx] = v;
}

// ---------------------------------------------------------------------------
// Tiled WMMA GEMM: Y[M x 1024] = A[M x 1024] @ W[1024 x 1024]
// Block tile 128(M) x 64(N), K-step 64 (2 WMMA K-steps per stage).
// 8 waves, wave tile 32x32 (2x2 WMMA). A row-major bf16 in LDS,
// W transposed in LDS -> all fragments are contiguous ds_load_b128 pairs.
// Panel staging is split gather->convert so all global loads issue as one
// clause (latency overlapped) instead of load/wait/store per chunk.
// Epilogue modes: 0 plain bf16 (head-major), 1 LN*scale bf16, 2 +bias f32.
// ---------------------------------------------------------------------------
__global__ __launch_bounds__(256) void gemm_proj(
    const float* __restrict__ A, const float* __restrict__ W, int M, int S,
    void* __restrict__ dstp, int mode, int Jdst, int joff,
    const float* __restrict__ g, const float* __restrict__ bvec, float scale,
    const float* __restrict__ bias) {
  __shared__ unsigned short As[128][72]; // 128 x 64(K) bf16, 16B-mult stride
  __shared__ unsigned short BsT[64][72]; // 64(N) x 64(K) bf16, transposed
  __shared__ float Cf[128][64];

  const int tid = threadIdx.x;
  const int lane = tid & 31, wave = tid >> 5;
  const int ln = lane & 15, hl = lane >> 4;
  const int wr = (wave >> 1) * 32;
  const int wc = (wave & 1) * 32;
  const int n0 = blockIdx.x * 64;
  const int m0 = blockIdx.y * 128;

  v8f acc[2][2];
#pragma unroll
  for (int tm = 0; tm < 2; ++tm)
#pragma unroll
    for (int tn = 0; tn < 2; ++tn) acc[tm][tn] = {};

  for (int k0 = 0; k0 < EMBD; k0 += 64) {
    // ---- gather: issue all panel loads up front (distinct registers) ----
    float4 da[8];
#pragma unroll
    for (int it = 0; it < 8; ++it) { // A panel: 128x64 = 2048 float4 / 256
      int idx = tid + it * 256;
      int r = idx >> 4, c4 = (idx & 15) * 4;
      da[it] = *(const float4*)&A[(size_t)(m0 + r) * EMBD + k0 + c4];
    }
    float4 dw[4];
#pragma unroll
    for (int it = 0; it < 4; ++it) { // W panel: 64x64 = 1024 float4 / 256
      int idx = tid + it * 256;
      int r = idx >> 4, c4 = (idx & 15) * 4;
      dw[it] = *(const float4*)&W[(size_t)(k0 + r) * EMBD + n0 + c4];
    }
    // prefetch next K panel toward L2 (global_prefetch_b8)
    if (k0 + 64 < EMBD) {
      __builtin_prefetch(
          &A[(size_t)(m0 + (tid >> 1)) * EMBD + k0 + 64 + (tid & 1) * 32], 0, 1);
      if (tid < 128)
        __builtin_prefetch(
            &W[(size_t)(k0 + 64 + (tid >> 1)) * EMBD + n0 + (tid & 1) * 32], 0,
            1);
    }

    __syncthreads(); // previous stage fully consumed

    // ---- convert + store to LDS ----
#pragma unroll
    for (int it = 0; it < 8; ++it) {
      int idx = tid + it * 256;
      int r = idx >> 4, c4 = (idx & 15) * 4;
      uint2 p;
      p.x = f2bf2(da[it].x, da[it].y);
      p.y = f2bf2(da[it].z, da[it].w);
      *(uint2*)&As[r][c4] = p;
    }
#pragma unroll
    for (int it = 0; it < 4; ++it) {
      int idx = tid + it * 256;
      int r = idx >> 4, c4 = (idx & 15) * 4;
      BsT[c4 + 0][r] = f2bf(dw[it].x);
      BsT[c4 + 1][r] = f2bf(dw[it].y);
      BsT[c4 + 2][r] = f2bf(dw[it].z);
      BsT[c4 + 3][r] = f2bf(dw[it].w);
    }
    __syncthreads();

#pragma unroll
    for (int ks = 0; ks < 2; ++ks) {
      FragAB a[2], b[2];
#pragma unroll
      for (int t = 0; t < 2; ++t) {
        int mrow = wr + t * 16 + ln;
        a[t].h[0] = *(const v8us*)&As[mrow][ks * 32 + hl * 8];
        a[t].h[1] = *(const v8us*)&As[mrow][ks * 32 + hl * 8 + 16];
      }
#pragma unroll
      for (int t = 0; t < 2; ++t) {
        int ncol = wc + t * 16 + ln;
        b[t].h[0] = *(const v8us*)&BsT[ncol][ks * 32 + hl * 8];
        b[t].h[1] = *(const v8us*)&BsT[ncol][ks * 32 + hl * 8 + 16];
      }
#pragma unroll
      for (int tm = 0; tm < 2; ++tm)
#pragma unroll
        for (int tn = 0; tn < 2; ++tn)
          acc[tm][tn] = __builtin_amdgcn_wmma_f32_16x16x32_bf16(
              false, a[tm].b, false, b[tn].b, (short)0, acc[tm][tn], false,
              false);
    }
  }

  __syncthreads();
#pragma unroll
  for (int tm = 0; tm < 2; ++tm)
#pragma unroll
    for (int tn = 0; tn < 2; ++tn)
#pragma unroll
      for (int r = 0; r < 8; ++r)
        Cf[wr + tm * 16 + hl * 8 + r][wc + tn * 16 + ln] = acc[tm][tn][r];
  __syncthreads();

  if (mode == 2) {
    float* out = (float*)dstp;
#pragma unroll
    for (int it = 0; it < 8; ++it) {
      int idx = tid + it * 256;
      int r = idx >> 4, c4 = (idx & 15) * 4;
      float4 o;
      o.x = Cf[r][c4 + 0] + bias[n0 + c4 + 0];
      o.y = Cf[r][c4 + 1] + bias[n0 + c4 + 1];
      o.z = Cf[r][c4 + 2] + bias[n0 + c4 + 2];
      o.w = Cf[r][c4 + 3] + bias[n0 + c4 + 3];
      *(float4*)&out[(size_t)(m0 + r) * EMBD + n0 + c4] = o;
    }
    return;
  }

  unsigned short* out = (unsigned short*)dstp;
  const int head = n0 >> 6;
  if (mode == 1) {
    if (tid < 128) {
      int r = tid;
      int gm = m0 + r;
      int bidx = gm / S, s = gm % S;
      size_t drow = ((size_t)(bidx * NHEADS + head) * Jdst + joff + s) * HSZ;
      float mean = 0.f;
      for (int c = 0; c < 64; ++c) mean += Cf[r][c];
      mean *= (1.f / 64.f);
      float var = 0.f;
      for (int c = 0; c < 64; ++c) {
        float d = Cf[r][c] - mean;
        var += d * d;
      }
      var *= (1.f / 64.f);
      float rs = rsqrtf(var + 1e-5f);
#pragma unroll
      for (int c8 = 0; c8 < 64; c8 += 8) {
        float y[8];
#pragma unroll
        for (int i = 0; i < 8; ++i)
          y[i] =
              ((Cf[r][c8 + i] - mean) * rs * g[c8 + i] + bvec[c8 + i]) * scale;
        uint4 pk;
        pk.x = f2bf2(y[0], y[1]);
        pk.y = f2bf2(y[2], y[3]);
        pk.z = f2bf2(y[4], y[5]);
        pk.w = f2bf2(y[6], y[7]);
        *(uint4*)&out[drow + c8] = pk;
      }
    }
  } else { // mode 0
#pragma unroll
    for (int it = 0; it < 4; ++it) {
      int idx = tid + it * 256;
      int r = idx >> 3, c8 = (idx & 7) * 8;
      int gm = m0 + r;
      int bidx = gm / S, s = gm % S;
      size_t drow = ((size_t)(bidx * NHEADS + head) * Jdst + joff + s) * HSZ;
      uint4 pk;
      pk.x = f2bf2(Cf[r][c8 + 0], Cf[r][c8 + 1]);
      pk.y = f2bf2(Cf[r][c8 + 2], Cf[r][c8 + 3]);
      pk.z = f2bf2(Cf[r][c8 + 4], Cf[r][c8 + 5]);
      pk.w = f2bf2(Cf[r][c8 + 6], Cf[r][c8 + 7]);
      *(uint4*)&out[drow + c8] = pk;
    }
  }
}

// ---------------------------------------------------------------------------
// Flash attention per (b,h): Q tile 128 rows, key tiles of 64, software-
// pipelined double-buffered K (async-to-LDS) and V (b128 + transpose store).
// Buffer 1 of K/V overlays the Q staging tile (Q is consumed into register
// fragments before the first loop barrier; first write into the overlay is
// issued only after that barrier). Softmax runs in base-2: log2(e) is folded
// into the q projection scale, so exponentials are raw v_exp_f32.
// ---------------------------------------------------------------------------
__global__ __launch_bounds__(256) void attn_kernel(
    const unsigned short* __restrict__ Q, const unsigned short* __restrict__ Kc,
    const unsigned short* __restrict__ Vc, const float* __restrict__ maskI,
    const float* __restrict__ maskJ, float* __restrict__ out) {
  __shared__ unsigned short pool[128 * 72]; // Qs; later K[1] (lo) + V[1] (hi)
  __shared__ unsigned short Ks0[64 * 72];
  __shared__ unsigned short Vs0[64 * 72];
  __shared__ unsigned short Ps[8][16][40];
  __shared__ float kmS[2][64];

  const int tid = threadIdx.x;
  const int lane = tid & 31, wave = tid >> 5;
  const int ln = lane & 15, hl = lane >> 4;
  const int bh = blockIdx.y;
  const int b = bh / NHEADS;
  const int h = bh % NHEADS;
  const int i0 = blockIdx.x * 128;
  const int qr = wave * 16;

  // ---- async-load Q tile into pool (row stride 72 shorts) ----
  const unsigned short* qsrc = Q + ((size_t)bh * SEQI + i0) * HSZ;
#pragma unroll
  for (int it = 0; it < 4; ++it) {
    int idx = tid + it * 256;
    int r = idx >> 3, c8 = (idx & 7) * 8;
    async_b128(&pool[r * 72 + c8], qsrc + (size_t)r * HSZ + c8);
  }
  wait_async0();
  __syncthreads();

  FragAB aq[2];
#pragma unroll
  for (int ks = 0; ks < 2; ++ks) {
    aq[ks].h[0] = *(const v8us*)&pool[(qr + ln) * 72 + ks * 32 + hl * 8];
    aq[ks].h[1] = *(const v8us*)&pool[(qr + ln) * 72 + ks * 32 + hl * 8 + 16];
  }

  float mrow[8];
#pragma unroll
  for (int r = 0; r < 8; ++r)
    mrow[r] = maskI[(size_t)b * SEQI + i0 + qr + hl * 8 + r];

  float m_i[8], l_i[8];
#pragma unroll
  for (int r = 0; r < 8; ++r) {
    m_i[r] = -3.0e38f;
    l_i[r] = 0.f;
  }
  v8f o[4];
#pragma unroll
  for (int td = 0; td < 4; ++td) o[td] = {};

  const unsigned short* ksrc = Kc + (size_t)bh * JTOT * HSZ;
  const unsigned short* vsrc = Vc + (size_t)bh * JTOT * HSZ;

  // ---- preload tile 0 into buffer 0 ----
  {
#pragma unroll
    for (int it = 0; it < 2; ++it) {
      int idx = tid + it * 256;
      int r = idx >> 3, c8 = (idx & 7) * 8;
      async_b128(&Ks0[r * 72 + c8], ksrc + (size_t)r * HSZ + c8);
      union { uint4 v; unsigned short us[8]; } d;
      d.v = *(const uint4*)(vsrc + (size_t)r * HSZ + c8);
#pragma unroll
      for (int i = 0; i < 8; ++i) Vs0[(c8 + i) * 72 + r] = d.us[i];
    }
    if (tid < 64) kmS[0][tid] = maskJ[(size_t)b * JTOT + tid];
    wait_async0();
  }

  int buf = 0;
  for (int jt = 0; jt < JTOT; jt += 64, buf ^= 1) {
    __syncthreads(); // tile(buf) staged & visible; buf^1 free for reuse

    unsigned short* Kcur = buf ? pool : Ks0;
    unsigned short* Vcur = buf ? (pool + 64 * 72) : Vs0;
    unsigned short* Knxt = buf ? Ks0 : pool;
    unsigned short* Vnxt = buf ? Vs0 : (pool + 64 * 72);

    const bool more = (jt + 64) < JTOT;
    uint4 vn[2];
    float kmn = 0.f;
    if (more) {
      // issue async K(i+1) and pull V(i+1)/mask(i+1) into registers, all
      // in flight while we compute tile i below
#pragma unroll
      for (int it = 0; it < 2; ++it) {
        int idx = tid + it * 256;
        int r = idx >> 3, c8 = (idx & 7) * 8;
        async_b128(&Knxt[r * 72 + c8],
                   ksrc + (size_t)(jt + 64 + r) * HSZ + c8);
        vn[it] = *(const uint4*)(vsrc + (size_t)(jt + 64 + r) * HSZ + c8);
      }
      if (tid < 64) kmn = maskJ[(size_t)b * JTOT + jt + 64 + tid];
    }

    // ---- S = Q K^T (wave tile 16 x 64), base-2 scaled ----
    v8f s[4];
#pragma unroll
    for (int tn = 0; tn < 4; ++tn) {
      s[tn] = {};
#pragma unroll
      for (int ks = 0; ks < 2; ++ks) {
        FragAB bk;
        bk.h[0] =
            *(const v8us*)&Kcur[(tn * 16 + ln) * 72 + ks * 32 + hl * 8];
        bk.h[1] =
            *(const v8us*)&Kcur[(tn * 16 + ln) * 72 + ks * 32 + hl * 8 + 16];
        s[tn] = __builtin_amdgcn_wmma_f32_16x16x32_bf16(
            false, aq[ks].b, false, bk.b, (short)0, s[tn], false, false);
      }
    }

    // ---- mask ----
#pragma unroll
    for (int tn = 0; tn < 4; ++tn) {
      float km = kmS[buf][tn * 16 + ln];
#pragma unroll
      for (int r = 0; r < 8; ++r)
        if (!(mrow[r] * km > 0.f)) s[tn][r] = -1e9f;
    }

    // ---- online softmax (base 2) ----
    float mnew[8], alpha[8];
#pragma unroll
    for (int r = 0; r < 8; ++r) {
      float v = fmaxf(fmaxf(s[0][r], s[1][r]), fmaxf(s[2][r], s[3][r]));
      v = fmaxf(v, __shfl_xor(v, 1, 32));
      v = fmaxf(v, __shfl_xor(v, 2, 32));
      v = fmaxf(v, __shfl_xor(v, 4, 32));
      v = fmaxf(v, __shfl_xor(v, 8, 32));
      mnew[r] = fmaxf(m_i[r], v);
      alpha[r] = fast_exp2(m_i[r] - mnew[r]);
      m_i[r] = mnew[r];
    }
    float rsum[8];
#pragma unroll
    for (int r = 0; r < 8; ++r) rsum[r] = 0.f;
#pragma unroll
    for (int tn = 0; tn < 4; ++tn)
#pragma unroll
      for (int r = 0; r < 8; ++r) {
        float p = fast_exp2(s[tn][r] - mnew[r]);
        s[tn][r] = p;
        rsum[r] += p;
      }
#pragma unroll
    for (int r = 0; r < 8; ++r) {
      float v = rsum[r];
      v += __shfl_xor(v, 1, 32);
      v += __shfl_xor(v, 2, 32);
      v += __shfl_xor(v, 4, 32);
      v += __shfl_xor(v, 8, 32);
      l_i[r] = l_i[r] * alpha[r] + v;
    }
#pragma unroll
    for (int td = 0; td < 4; ++td)
#pragma unroll
      for (int r = 0; r < 8; ++r) o[td][r] *= alpha[r];

    // ---- P -> LDS (wave-private, frag-ready rows) ----
#pragma unroll
    for (int tn = 0; tn < 4; ++tn)
#pragma unroll
      for (int r = 0; r < 8; ++r)
        Ps[wave][hl * 8 + r][tn * 16 + ln] = f2bf(s[tn][r]);

    // ---- O += P @ V ----
#pragma unroll
    for (int ks = 0; ks < 2; ++ks) {
      FragAB ap;
      ap.h[0] = *(const v8us*)&Ps[wave][ln][ks * 32 + hl * 8];
      ap.h[1] = *(const v8us*)&Ps[wave][ln][ks * 32 + hl * 8 + 16];
#pragma unroll
      for (int td = 0; td < 4; ++td) {
        FragAB bv;
        bv.h[0] =
            *(const v8us*)&Vcur[(td * 16 + ln) * 72 + ks * 32 + hl * 8];
        bv.h[1] =
            *(const v8us*)&Vcur[(td * 16 + ln) * 72 + ks * 32 + hl * 8 + 16];
        o[td] = __builtin_amdgcn_wmma_f32_16x16x32_bf16(
            false, ap.b, false, bv.b, (short)0, o[td], false, false);
      }
    }

    // ---- stage tile i+1 V (transpose) + wait for K(i+1) ----
    if (more) {
#pragma unroll
      for (int it = 0; it < 2; ++it) {
        int idx = tid + it * 256;
        int r = idx >> 3, c8 = (idx & 7) * 8;
        union { uint4 v; unsigned short us[8]; } d;
        d.v = vn[it];
#pragma unroll
        for (int i = 0; i < 8; ++i) Vnxt[(c8 + i) * 72 + r] = d.us[i];
      }
      if (tid < 64) kmS[buf ^ 1][tid] = kmn;
      wait_async0();
    }
  }

  // ---- finalize: divide by l, write f32 attnout[b][i][h*64+d] ----
#pragma unroll
  for (int r = 0; r < 8; ++r) {
    float inv = (l_i[r] > 0.f) ? (1.f / l_i[r]) : 0.f;
    int i = i0 + qr + hl * 8 + r;
    size_t row = ((size_t)b * SEQI + i) * EMBD + h * HSZ;
#pragma unroll
    for (int td = 0; td < 4; ++td) out[row + td * 16 + ln] = o[td][r] * inv;
  }
}

// ---------------------------------------------------------------------------
// Launch
// ---------------------------------------------------------------------------
extern "C" void kernel_launch(void* const* d_in, const int* in_sizes, int n_in,
                              void* d_out, int out_size, void* d_ws,
                              size_t ws_size, hipStream_t stream) {
  const float* x = (const float*)d_in[0];
  const float* mm0 = (const float*)d_in[1];
  const float* mm1 = (const float*)d_in[2];
  const float* mask = (const float*)d_in[3];
  const float* mmm0 = (const float*)d_in[4];
  const float* mmm1 = (const float*)d_in[5];
  const float* Wq = (const float*)d_in[6];
  const float* Wk = (const float*)d_in[7];
  const float* Wv = (const float*)d_in[8];
  const float* Wu = (const float*)d_in[9];
  const float* bu = (const float*)d_in[10];
  const float* qg = (const float*)d_in[11];
  const float* qb = (const float*)d_in[12];
  const float* kg = (const float*)d_in[13];
  const float* kb = (const float*)d_in[14];
  const float* mmWk0 = (const float*)d_in[15];
  const float* mmWv0 = (const float*)d_in[16];
  const float* mmkg0 = (const float*)d_in[17];
  const float* mmkb0 = (const float*)d_in[18];
  const float* mmWk1 = (const float*)d_in[19];
  const float* mmWv1 = (const float*)d_in[20];
  const float* mmkg1 = (const float*)d_in[21];
  const float* mmkb1 = (const float*)d_in[22];

  char* ws = (char*)d_ws;
  unsigned short* q_ws = (unsigned short*)(ws);               // 4 MB
  unsigned short* k_ws = (unsigned short*)(ws + (4u << 20));  // 8 MB
  unsigned short* v_ws = (unsigned short*)(ws + (12u << 20)); // 8 MB
  float* attn_ws = (float*)(ws + (20u << 20));                // 8 MB
  float* kmask_ws = (float*)(ws + (28u << 20));               // 16 KB

  const float invs = 0.17677669529663687f; // EMB^(-0.25)
  const float l2e = 1.4426950408889634f;   // log2(e), folded into q only

  build_mask<<<dim3((NB * JTOT + 255) / 256), dim3(256), 0, stream>>>(
      mask, mmm0, mmm1, kmask_ws);

  dim3 blk(256);
  dim3 gx(EMBD / 64, (NB * SEQI) / 128); // (16,16)
  dim3 gm(EMBD / 64, (NB * SEQC) / 128); // (16,8)

  gemm_proj<<<gx, blk, 0, stream>>>(x, Wq, NB * SEQI, SEQI, q_ws, 1, SEQI, 0,
                                    qg, qb, invs * l2e, nullptr);
  gemm_proj<<<gx, blk, 0, stream>>>(x, Wk, NB * SEQI, SEQI, k_ws, 1, JTOT, 0,
                                    kg, kb, invs, nullptr);
  gemm_proj<<<gx, blk, 0, stream>>>(x, Wv, NB * SEQI, SEQI, v_ws, 0, JTOT, 0,
                                    nullptr, nullptr, 1.f, nullptr);
  gemm_proj<<<gm, blk, 0, stream>>>(mm0, mmWk0, NB * SEQC, SEQC, k_ws, 1, JTOT,
                                    SEQI, mmkg0, mmkb0, invs, nullptr);
  gemm_proj<<<gm, blk, 0, stream>>>(mm0, mmWv0, NB * SEQC, SEQC, v_ws, 0, JTOT,
                                    SEQI, nullptr, nullptr, 1.f, nullptr);
  gemm_proj<<<gm, blk, 0, stream>>>(mm1, mmWk1, NB * SEQC, SEQC, k_ws, 1, JTOT,
                                    SEQI + SEQC, mmkg1, mmkb1, invs, nullptr);
  gemm_proj<<<gm, blk, 0, stream>>>(mm1, mmWv1, NB * SEQC, SEQC, v_ws, 0, JTOT,
                                    SEQI + SEQC, nullptr, nullptr, 1.f, nullptr);

  attn_kernel<<<dim3(SEQI / 128, NB * NHEADS), blk, 0, stream>>>(
      q_ws, k_ws, v_ws, mask, kmask_ws, attn_ws);

  gemm_proj<<<gx, blk, 0, stream>>>(attn_ws, Wu, NB * SEQI, SEQI, d_out, 2, 0,
                                    0, nullptr, nullptr, 1.f, bu);
}
